// LittleBitLinearMultiRankHF_29480655520419
// MI455X (gfx1250) — matrix operand order repaired and checked by reading, not compiled, not run on hardware
//
#include <hip/hip_runtime.h>
#include <math.h>

// Problem dims (fixed by the reference)
#define D_IN   4096
#define D_OUT  4096
#define RANK   256
#define KSMALL 16
#define MROWS  8192   // B * S = 4 * 2048

typedef float        f32x4 __attribute__((ext_vector_type(4)));
typedef unsigned int u32x4 __attribute__((ext_vector_type(4)));
typedef float        v8f   __attribute__((ext_vector_type(8)));
typedef __bf16       v16bf __attribute__((ext_vector_type(16)));

union Frag {            // 32 bytes: two b128 LDS loads -> one WMMA operand
  u32x4 q[2];
  v16bf v;
};

__device__ __forceinline__ unsigned short f2bf(float f) {
  unsigned int u = __builtin_bit_cast(unsigned int, f);
  u += 0x7FFFu + ((u >> 16) & 1u);          // round-to-nearest-even
  return (unsigned short)(u >> 16);
}

// gfx1250 async global->LDS copy, 16 bytes per lane, tracked by ASYNCcnt.
// vdst = LDS byte offset (low 32 bits of the generic LDS pointer), vaddr = 64b global addr.
__device__ __forceinline__ void async_b128(const void* g, const unsigned short* l) {
  unsigned int loff = (unsigned int)(unsigned long long)l;
  asm volatile("global_load_async_to_lds_b128 %0, %1, off"
               :: "v"(loff), "v"(g)
               : "memory");
}

__device__ __forceinline__ void wait_async0() {
#if __has_builtin(__builtin_amdgcn_s_wait_asynccnt)
  __builtin_amdgcn_s_wait_asynccnt(0);
#else
  asm volatile("s_wait_asynccnt 0" ::: "memory");
#endif
}

// ---------------------------------------------------------------------------
// Stage 1: Up[o][j] = tanh(100*U_fp) * (U_mag @ V_mag_u)   (bf16, row-major)
//          VpT[j][i] = tanh(100*V_fp) * (V_mag_g @ V_mag_lv) (bf16, TRANSPOSED)
// ---------------------------------------------------------------------------
__global__ __launch_bounds__(256)
void k_factors(const float* __restrict__ U_fp, const float* __restrict__ U_mag,
               const float* __restrict__ V_mag_u,
               const float* __restrict__ V_fp, const float* __restrict__ V_mag_g,
               const float* __restrict__ V_mag_lv,
               unsigned short* __restrict__ Up,    // [D_OUT][RANK]
               unsigned short* __restrict__ VpT)   // [RANK][D_IN]
{
  const int idx = blockIdx.x * 256 + threadIdx.x;   // 0 .. D*RANK-1
  if (blockIdx.y == 0) {
    const int j = idx & (RANK - 1);
    const int o = idx >> 8;
    float acc = 0.f;
#pragma unroll
    for (int k = 0; k < KSMALL; ++k)
      acc += U_mag[o * KSMALL + k] * V_mag_u[k * RANK + j];
    const float s = tanhf(100.0f * U_fp[o * RANK + j]);
    Up[o * RANK + j] = f2bf(s * acc);
  } else {
    const int i = idx & (D_IN - 1);                 // fast index -> coalesced VpT store
    const int j = idx >> 12;
    float acc = 0.f;
#pragma unroll
    for (int k = 0; k < KSMALL; ++k)
      acc += V_mag_g[i * KSMALL + k] * V_mag_lv[k * RANK + j];
    const float s = tanhf(100.0f * V_fp[i * RANK + j]);
    VpT[j * D_IN + i] = f2bf(s * acc);
  }
}

// ---------------------------------------------------------------------------
// Stage 2: T[m][n] = sum_k x[m][k] * Vp[k][n]   (M=8192, N=256, K=4096)
// 256 threads (8 waves), tile 64(M) x 256(N), BK=32, double-buffered LDS.
// B tile (bf16): async global->LDS.  A tile (fp32->bf16): reg prefetch+convert.
// Inner fragment loop is software-pipelined with 2 B-fragment register sets.
// ---------------------------------------------------------------------------
#define S2_BM 64
#define S2_BK 32
#define S2_BS 40   // padded LDS stride (bf16 elems): 80B rows, keeps 16B align

__global__ __launch_bounds__(256)
void k_xv(const float* __restrict__ x,             // [MROWS][D_IN] fp32
          const unsigned short* __restrict__ VpT,  // [RANK][D_IN] bf16
          unsigned short* __restrict__ T)          // [MROWS][RANK] bf16
{
  __shared__ unsigned short sA[2][S2_BM * S2_BK];   // x tile bf16, 2 x 4 KB
  __shared__ unsigned short sB[2][RANK * S2_BS];    // Vp^T tile, 2 x 20 KB

  const int tid     = threadIdx.x;
  const int wave    = tid >> 5;
  const int lane    = tid & 31;
  const int lane_lo = lane & 15;
  const int lhi     = lane >> 4;
  const int m0      = blockIdx.x * S2_BM;
  const int mt      = wave & 3;
  const int nb      = (wave >> 2) * 8;

  // Per-thread copy coordinates (fixed across k-steps)
  const int a_row0 = tid >> 3;                 // chunk 0: rows 0..31
  const int a_col  = (tid & 7) * 4;
  const int b_n    = tid >> 2;                 // chunk i: n = b_n + i*64
  const int b_c    = (tid & 3) * 8;

  v8f acc[8] = {};

  // ---- prologue: tile 0 ----
  {
#pragma unroll
    for (int i = 0; i < 4; ++i)
      async_b128(VpT + (size_t)(b_n + i * 64) * D_IN + b_c,
                 &sB[0][(b_n + i * 64) * S2_BS + b_c]);
#pragma unroll
    for (int i = 0; i < 2; ++i) {
      const int row = a_row0 + i * 32;
      f32x4 v = *(const f32x4*)(x + (size_t)(m0 + row) * D_IN + a_col);
      sA[0][row * S2_BK + a_col + 0] = f2bf(v.x);
      sA[0][row * S2_BK + a_col + 1] = f2bf(v.y);
      sA[0][row * S2_BK + a_col + 2] = f2bf(v.z);
      sA[0][row * S2_BK + a_col + 3] = f2bf(v.w);
    }
    wait_async0();
    __syncthreads();
  }

  const int NK = D_IN / S2_BK;   // 128
  for (int kt = 0; kt < NK; ++kt) {
    const int p    = kt & 1;
    const bool more = (kt + 1 < NK);
    f32x4 r[2];

    if (more) {
      const int k1 = (kt + 1) * S2_BK;
      // async B(kt+1) -> sB[1-p]; lands any time before the end-of-iter wait
#pragma unroll
      for (int i = 0; i < 4; ++i)
        async_b128(VpT + (size_t)(b_n + i * 64) * D_IN + k1 + b_c,
                   &sB[1 - p][(b_n + i * 64) * S2_BS + b_c]);
      // A(kt+1) register prefetch: latency hides behind the 8 WMMAs below
#pragma unroll
      for (int i = 0; i < 2; ++i)
        r[i] = *(const f32x4*)(x + (size_t)(m0 + a_row0 + i * 32) * D_IN + k1 + a_col);
    }

    // ---- compute on buffer p ----
    Frag fa;
    const int abase = (mt * 16 + lane_lo) * S2_BK + lhi * 16;
    fa.q[0] = *(const u32x4*)(&sA[p][abase]);
    fa.q[1] = *(const u32x4*)(&sA[p][abase + 8]);

    const int bb0 = lane_lo * S2_BS + lhi * 16;   // per-lane base within an n-tile

    // Software-pipelined: fb[t+1] loads issue before wmma(t) so ds latency
    // hides behind the previous matrix op (no dscnt==0 stall per WMMA).
    Frag fb[2];
    fb[0].q[0] = *(const u32x4*)(&sB[p][(nb + 0) * 16 * S2_BS + bb0]);
    fb[0].q[1] = *(const u32x4*)(&sB[p][(nb + 0) * 16 * S2_BS + bb0 + 8]);
#pragma unroll
    for (int t = 0; t < 8; ++t) {
      if (t < 7) {
        const int bbase = (nb + t + 1) * 16 * S2_BS + bb0;
        fb[(t + 1) & 1].q[0] = *(const u32x4*)(&sB[p][bbase]);
        fb[(t + 1) & 1].q[1] = *(const u32x4*)(&sB[p][bbase + 8]);
      }
      acc[t] = __builtin_amdgcn_wmma_f32_16x16x32_bf16(
                   false, fa.v, false, fb[t & 1].v, (short)0, acc[t], false, false);
    }

    if (more) {
#pragma unroll
      for (int i = 0; i < 2; ++i) {
        const int row = a_row0 + i * 32;
        sA[1 - p][row * S2_BK + a_col + 0] = f2bf(r[i].x);
        sA[1 - p][row * S2_BK + a_col + 1] = f2bf(r[i].y);
        sA[1 - p][row * S2_BK + a_col + 2] = f2bf(r[i].z);
        sA[1 - p][row * S2_BK + a_col + 3] = f2bf(r[i].w);
      }
    }
    wait_async0();
    __syncthreads();
  }

  // C/D layout: lane l, vgpr v -> (M = v + 8*(l>=16), N = l&15)
#pragma unroll
  for (int t = 0; t < 8; ++t) {
    const int n = (nb + t) * 16 + lane_lo;
#pragma unroll
    for (int v = 0; v < 8; ++v) {
      const int m = m0 + mt * 16 + v + 8 * lhi;
      T[(size_t)m * RANK + n] = f2bf(acc[t][v]);
    }
  }
}

// ---------------------------------------------------------------------------
// Stage 3: y[m][o] = sum_j T[m][j] * Up[o][j] + bias[o]  (M=8192, N=4096, K=256)
// 256 threads (8 waves), tile 128(M) x 128(N), BK=32, fully async double-buffered.
// ---------------------------------------------------------------------------
#define S3_BM 128
#define S3_BN 128
#define S3_BK 32

__global__ __launch_bounds__(256)
void k_out(const unsigned short* __restrict__ T,    // [MROWS][RANK] bf16
           const unsigned short* __restrict__ Up,   // [D_OUT][RANK] bf16
           const float* __restrict__ bias,          // [D_OUT]
           float* __restrict__ y)                   // [MROWS][D_OUT] fp32
{
  __shared__ unsigned short sA[2][S3_BM * S3_BK];   // 2 x 8 KB
  __shared__ unsigned short sB[2][S3_BN * S3_BK];   // 2 x 8 KB

  const int tid     = threadIdx.x;
  const int wave    = tid >> 5;
  const int lane    = tid & 31;
  const int lane_lo = lane & 15;
  const int lhi     = lane >> 4;
  const int m0      = blockIdx.x * S3_BM;
  const int n0      = blockIdx.y * S3_BN;

  const int c_r = tid >> 2;            // chunk i: row = c_r + i*64
  const int c_c = (tid & 3) * 8;

  v8f acc[8] = {};

  // ---- prologue: tile 0 ----
#pragma unroll
  for (int i = 0; i < 2; ++i) {
    const int r = c_r + i * 64;
    async_b128(T  + (size_t)(m0 + r) * RANK + c_c, &sA[0][r * S3_BK + c_c]);
    async_b128(Up + (size_t)(n0 + r) * RANK + c_c, &sB[0][r * S3_BK + c_c]);
  }
  wait_async0();
  __syncthreads();

  const int NK = RANK / S3_BK;   // 8
  for (int kt = 0; kt < NK; ++kt) {
    const int p    = kt & 1;
    const bool more = (kt + 1 < NK);

    if (more) {
      const int k1 = (kt + 1) * S3_BK;
#pragma unroll
      for (int i = 0; i < 2; ++i) {
        const int r = c_r + i * 64;
        async_b128(T  + (size_t)(m0 + r) * RANK + k1 + c_c, &sA[1 - p][r * S3_BK + c_c]);
        async_b128(Up + (size_t)(n0 + r) * RANK + k1 + c_c, &sB[1 - p][r * S3_BK + c_c]);
      }
    }

    Frag fa;
    const int abase = (wave * 16 + lane_lo) * S3_BK + lhi * 16;
    fa.q[0] = *(const u32x4*)(&sA[p][abase]);
    fa.q[1] = *(const u32x4*)(&sA[p][abase + 8]);

    const int bb0 = lane_lo * S3_BK + lhi * 16;

    Frag fb[2];
    fb[0].q[0] = *(const u32x4*)(&sB[p][bb0]);
    fb[0].q[1] = *(const u32x4*)(&sB[p][bb0 + 8]);
#pragma unroll
    for (int t = 0; t < 8; ++t) {
      if (t < 7) {
        const int bbase = (t + 1) * 16 * S3_BK + bb0;
        fb[(t + 1) & 1].q[0] = *(const u32x4*)(&sB[p][bbase]);
        fb[(t + 1) & 1].q[1] = *(const u32x4*)(&sB[p][bbase + 8]);
      }
      acc[t] = __builtin_amdgcn_wmma_f32_16x16x32_bf16(
                   false, fa.v, false, fb[t & 1].v, (short)0, acc[t], false, false);
    }

    wait_async0();
    __syncthreads();
  }

#pragma unroll
  for (int t = 0; t < 8; ++t) {
    const int n = n0 + t * 16 + lane_lo;
    const float b = bias[n];
#pragma unroll
    for (int v = 0; v < 8; ++v) {
      const int m = m0 + wave * 16 + v + 8 * lhi;
      y[(size_t)m * D_OUT + n] = acc[t][v] + b;
    }
  }
}

// ---------------------------------------------------------------------------
extern "C" void kernel_launch(void* const* d_in, const int* in_sizes, int n_in,
                              void* d_out, int out_size, void* d_ws, size_t ws_size,
                              hipStream_t stream) {
  const float* x        = (const float*)d_in[0];
  const float* U_fp     = (const float*)d_in[1];
  const float* V_fp     = (const float*)d_in[2];
  const float* U_mag    = (const float*)d_in[3];
  const float* V_mag_u  = (const float*)d_in[4];
  const float* V_mag_g  = (const float*)d_in[5];
  const float* V_mag_lv = (const float*)d_in[6];
  const float* bias     = (const float*)d_in[7];
  float* y = (float*)d_out;

  // Workspace carve-out (bf16): Up 2MB | VpT 2MB | T 4MB  = 8 MB total
  unsigned short* Up  = (unsigned short*)d_ws;
  unsigned short* VpT = Up  + (size_t)D_OUT * RANK;
  unsigned short* T   = VpT + (size_t)RANK * D_IN;

  k_factors<<<dim3((D_OUT * RANK) / 256, 2), 256, 0, stream>>>(
      U_fp, U_mag, V_mag_u, V_fp, V_mag_g, V_mag_lv, Up, VpT);

  k_xv<<<dim3(MROWS / S2_BM), 256, 0, stream>>>(x, VpT, T);

  k_out<<<dim3(MROWS / S3_BM, D_OUT / S3_BN), 256, 0, stream>>>(T, Up, bias, y);
}
